// SwinFeatureExtractor_47407849013788
// MI455X (gfx1250) — compile-verified, matrix operand order checked by reading, not run on hardware
//
#include <hip/hip_runtime.h>
#include <hip/hip_bf16.h>
#include <math.h>

typedef __attribute__((ext_vector_type(16))) _Float16 v16h;
typedef __attribute__((ext_vector_type(8)))  _Float16 v8h;
typedef __attribute__((ext_vector_type(8)))  float    v8f;

#define WS7 7
#define NHEAD 8
#define HDIM 32
#define CDIM 256
#define NTOK 49
#define HW 196
#define MROWS 50176           // 256*196
#define ATT_SCALE 0.1767766952966369f

// ---------------- WMMA helpers (CDNA5 gfx1250, wave32) ----------------

__device__ inline v8f wmma_f32(v16h a, v16h b, v8f c) {
  // D = A(16x32 f16) * B(32x16 f16) + C(16x16 f32)
  return __builtin_amdgcn_wmma_f32_16x16x32_f16(false, a, false, b, (short)0, c,
                                                false, false);
}

// 16 halves as two contiguous 8-half (16B) vector loads: halves [0..7] from p,
// halves [8..15] from p+16 (the K pattern's upper-VGPR run).
__device__ inline v16h ld16(const _Float16* p) {
  v8h lo = *(const v8h*)p;
  v8h hi = *(const v8h*)(p + 16);
  v16h f;
#pragma unroll
  for (int i = 0; i < 8; ++i) { f[i] = lo[i]; f[8 + i] = hi[i]; }
  return f;
}

__device__ inline v16h zero_v16h() {
  v16h f;
#pragma unroll
  for (int i = 0; i < 16; ++i) f[i] = (_Float16)0.f;
  return f;
}

// A fragment 16x32 f16 from row-major (row, K-contig) buffer, row = lane%16,
// kb = 8*(lane>=16); per ISA 16-bit A layout. B fragments use the SAME loader
// on an N-major (col, K-contig) staged tile (layout mirrored over N).
__device__ inline v16h frag_a(const _Float16* base, int stride, int lane) {
  return ld16(base + (lane & 15) * stride + ((lane >> 4) << 3));
}

// ---------------- Weight prep: f32 -> f16, optional transpose to N x K ----------------
__global__ __launch_bounds__(256) void k_cvt(const float* __restrict__ in,
                                             _Float16* __restrict__ out, int n) {
  int i = blockIdx.x * 256 + threadIdx.x;
  if (i < n) out[i] = (_Float16)in[i];
}
__global__ __launch_bounds__(256) void k_cvt_t(const float* __restrict__ in,
                                               _Float16* __restrict__ out,
                                               int K, int N) {
  // out[n*K + k] = in[k*N + n]
  int i = blockIdx.x * 256 + threadIdx.x;
  if (i < K * N) {
    int nrow = i / K, k = i - nrow * K;
    out[i] = (_Float16)in[(size_t)k * N + nrow];
  }
}

// ---------------- Kernel 1: 1x1 conv (x is B,C,H,W channel-major) ----------------
// tok[m, o] = sum_c conv_w[o, c] * x[b, c, hw] + conv_b[o],  m = b*196 + hw
__global__ __launch_bounds__(256) void k_conv(const float* __restrict__ x,
                                              const _Float16* __restrict__ cw16,  // N x K (=o x c)
                                              const float* __restrict__ cb,
                                              float* __restrict__ tok) {
  __shared__ _Float16 As[128 * 40];
  __shared__ _Float16 Bs[64 * 40];
  const int tid = threadIdx.x;
  const int m0 = blockIdx.x * 128, n0 = blockIdx.y * 64;
  const int wave = tid >> 5, lane = tid & 31;
  const int wm = (wave & 3) * 32, wn = (wave >> 2) * 32;
  v8f acc[2][2] = {};

  const int ar = tid >> 1, akk = (tid & 1) * 16;
  const int m = m0 + ar;
  const int bimg = m / HW, hw = m % HW;
  const size_t xbase = (size_t)bimg * (CDIM * HW) + hw;
  const int bn = tid >> 2, bkk = (tid & 3) * 8;
  const _Float16* bSrc = cw16 + (size_t)(n0 + bn) * CDIM + bkk;

  for (int k0 = 0; k0 < CDIM; k0 += 32) {
    __syncthreads();
    {  // A: 128x32 strided gather from x, fp32 -> f16 (transpose through LDS)
      _Float16* dst = As + ar * 40 + akk;
#pragma unroll
      for (int i = 0; i < 16; ++i)
        dst[i] = (_Float16)x[xbase + (size_t)(k0 + akk + i) * HW];
    }
    // B: already f16 N x K -> contiguous 16B copy
    *(v8h*)(Bs + bn * 40 + bkk) = *(const v8h*)(bSrc + k0);
    if (k0 + 32 < CDIM) __builtin_prefetch(bSrc + k0 + 32, 0, 3);
    __syncthreads();
    v16h a0 = frag_a(As + wm * 40, 40, lane);
    v16h a1 = frag_a(As + (wm + 16) * 40, 40, lane);
    v16h b0 = frag_a(Bs + wn * 40, 40, lane);
    v16h b1 = frag_a(Bs + (wn + 16) * 40, 40, lane);
    acc[0][0] = wmma_f32(a0, b0, acc[0][0]);
    acc[0][1] = wmma_f32(a0, b1, acc[0][1]);
    acc[1][0] = wmma_f32(a1, b0, acc[1][0]);
    acc[1][1] = wmma_f32(a1, b1, acc[1][1]);
  }
  const int cr = (lane >> 4) * 8, cc = lane & 15;
#pragma unroll
  for (int ti = 0; ti < 2; ++ti)
#pragma unroll
    for (int tj = 0; tj < 2; ++tj)
#pragma unroll
      for (int j = 0; j < 8; ++j) {
        int gr = m0 + wm + ti * 16 + cr + j;
        int gc = n0 + wn + tj * 16 + cc;
        tok[(size_t)gr * CDIM + gc] = acc[ti][tj][j] + cb[gc];
      }
}

// ---------------- Kernel 2: LayerNorm (+ optional window partition), f32 -> f16 ----------------
__global__ __launch_bounds__(256) void k_ln(const float* __restrict__ src,
                                            const float* __restrict__ g,
                                            const float* __restrict__ bb,
                                            _Float16* __restrict__ dst,
                                            int reorder) {
  const int wave = threadIdx.x >> 5, lane = threadIdx.x & 31;
  const int m = blockIdx.x * 8 + wave;
  const float* p = src + (size_t)m * CDIM;
  float v[8];
  float s = 0.f;
#pragma unroll
  for (int i = 0; i < 8; ++i) { v[i] = p[lane + i * 32]; s += v[i]; }
#pragma unroll
  for (int msk = 16; msk; msk >>= 1) s += __shfl_xor(s, msk, 32);
  const float mu = s * (1.f / 256.f);
  float vs = 0.f;
#pragma unroll
  for (int i = 0; i < 8; ++i) { float d = v[i] - mu; vs += d * d; }
#pragma unroll
  for (int msk = 16; msk; msk >>= 1) vs += __shfl_xor(vs, msk, 32);
  const float rstd = rsqrtf(vs * (1.f / 256.f) + 1e-5f);

  int mo = m;
  if (reorder) {  // token (b,h,w) -> windowed row: win = b*4 + wi*2 + wj, loc = li*7+lj
    int b = m / HW, hw = m % HW;
    int h = hw / 14, w = hw % 14;
    int wi = h / WS7, li = h % WS7, wj = w / WS7, lj = w % WS7;
    mo = (b * 4 + wi * 2 + wj) * NTOK + li * WS7 + lj;
  }
  _Float16* q = dst + (size_t)mo * CDIM;
#pragma unroll
  for (int i = 0; i < 8; ++i) {
    int c = lane + i * 32;
    q[c] = (_Float16)((v[i] - mu) * rstd * g[c] + bb[c]);
  }
}

// ---------------- Generic tiled WMMA GEMM: f16 A (MxK) x f16 Bt (NxK) + bias ----------------
template <class Epi>
__global__ __launch_bounds__(256) void k_gemm(const _Float16* __restrict__ A,
                                              const _Float16* __restrict__ Bt,
                                              const float* __restrict__ bias,
                                              int K, Epi epi) {
  __shared__ _Float16 As[128 * 40];
  __shared__ _Float16 Bs[64 * 40];
  const int tid = threadIdx.x;
  const int m0 = blockIdx.x * 128, n0 = blockIdx.y * 64;
  const int wave = tid >> 5, lane = tid & 31;
  const int wm = (wave & 3) * 32, wn = (wave >> 2) * 32;
  v8f acc[2][2] = {};

  const int ar = tid >> 1, akk = (tid & 1) * 16;
  const int bn = tid >> 2, bkk = (tid & 3) * 8;
  const _Float16* aSrc = A + (size_t)(m0 + ar) * K + akk;
  const _Float16* bSrc = Bt + (size_t)(n0 + bn) * K + bkk;

  for (int k0 = 0; k0 < K; k0 += 32) {
    __syncthreads();
    {  // A: two 16B vector copies per thread
      _Float16* dst = As + ar * 40 + akk;
      *(v8h*)dst       = *(const v8h*)(aSrc + k0);
      *(v8h*)(dst + 8) = *(const v8h*)(aSrc + k0 + 8);
    }
    // B: one 16B vector copy per thread
    *(v8h*)(Bs + bn * 40 + bkk) = *(const v8h*)(bSrc + k0);
    if (k0 + 32 < K) {  // prefetch next k-tile (global_prefetch_b8)
      __builtin_prefetch(aSrc + k0 + 32, 0, 3);
      __builtin_prefetch(bSrc + k0 + 32, 0, 3);
    }
    __syncthreads();
    v16h a0 = frag_a(As + wm * 40, 40, lane);
    v16h a1 = frag_a(As + (wm + 16) * 40, 40, lane);
    v16h b0 = frag_a(Bs + wn * 40, 40, lane);
    v16h b1 = frag_a(Bs + (wn + 16) * 40, 40, lane);
    acc[0][0] = wmma_f32(a0, b0, acc[0][0]);
    acc[0][1] = wmma_f32(a0, b1, acc[0][1]);
    acc[1][0] = wmma_f32(a1, b0, acc[1][0]);
    acc[1][1] = wmma_f32(a1, b1, acc[1][1]);
  }
  const int cr = (lane >> 4) * 8, cc = lane & 15;
#pragma unroll
  for (int ti = 0; ti < 2; ++ti)
#pragma unroll
    for (int tj = 0; tj < 2; ++tj)
#pragma unroll
      for (int j = 0; j < 8; ++j) {
        int gr = m0 + wm + ti * 16 + cr + j;
        int gc = n0 + wn + tj * 16 + cc;
        epi(gr, gc, acc[ti][tj][j] + bias[gc]);
      }
}

struct EpiF16 {  // plain f16 store (qkv)
  _Float16* out; int ldo;
  __device__ void operator()(int r, int c, float v) const {
    out[(size_t)r * ldo + c] = (_Float16)v;
  }
};
struct EpiGelu {  // exact GELU, f16 store (fc1)
  _Float16* out; int ldo;
  __device__ void operator()(int r, int c, float v) const {
    out[(size_t)r * ldo + c] = (_Float16)(0.5f * v * (1.f + erff(v * 0.70710678118f)));
  }
};
struct EpiProj {  // window-reverse + residual (proj): rows are windowed
  const float* tok; float* tok2;
  __device__ void operator()(int mw, int c, float v) const {
    int win = mw / NTOK, loc = mw % NTOK;
    int b = win >> 2, wr = win & 3;
    int li = loc / WS7, lj = loc % WS7;
    int h = (wr >> 1) * WS7 + li, w = (wr & 1) * WS7 + lj;
    size_t o = (size_t)(b * HW + h * 14 + w) * CDIM + c;
    tok2[o] = tok[o] + v;
  }
};
struct EpiFc2 {  // residual add (fc2)
  const float* tok2; float* tok3;
  __device__ void operator()(int r, int c, float v) const {
    size_t o = (size_t)r * CDIM + c;
    tok3[o] = tok2[o] + v;
  }
};

// ---------------- Kernel: windowed attention, 1 block per window, 1 wave per head ----------------
__global__ __launch_bounds__(256) void k_attn(const _Float16* __restrict__ qkv,
                                              const float* __restrict__ rpb,
                                              _Float16* __restrict__ obuf) {
  __shared__ _Float16 pl[NHEAD][16 * 72];  // per-wave P slice (16 x 64, padded)
  const int head = threadIdx.x >> 5, lane = threadIdx.x & 31;
  const int win = blockIdx.x;
  const int halfsel = lane >> 4, kb = halfsel * 8, nl = lane & 15;
  const size_t rowbase = (size_t)win * NTOK * 768;
  const int colq = head * HDIM, colk = 256 + head * HDIM, colv = 512 + head * HDIM;

  // K matrix as 4 B-fragments: B[d][n] = K[n][d], n padded 49 -> 64 (vector loads)
  v16h bk[4];
#pragma unroll
  for (int nt = 0; nt < 4; ++nt) {
    int n = nt * 16 + nl;
    bk[nt] = (n < NTOK) ? ld16(qkv + rowbase + (size_t)n * 768 + colk + kb)
                        : zero_v16h();
  }
  // V matrix as B-fragments over padded K=64: bv[kt][c16], B[kk][d] (row-strided)
  v16h bv[2][2];
#pragma unroll
  for (int kt = 0; kt < 2; ++kt)
#pragma unroll
    for (int c16 = 0; c16 < 2; ++c16) {
      v16h f;
      int d = c16 * 16 + nl;
#pragma unroll
      for (int i = 0; i < 8; ++i) {
        int kkA = (i < 4) ? (kb + 2 * i) : (16 + kb + 2 * (i - 4));
        int kk0 = kt * 32 + kkA;
        f[2 * i]     = (kk0 < NTOK)     ? qkv[rowbase + (size_t)kk0 * 768 + colv + d] : (_Float16)0.f;
        f[2 * i + 1] = (kk0 + 1 < NTOK) ? qkv[rowbase + (size_t)(kk0 + 1) * 768 + colv + d] : (_Float16)0.f;
      }
      bv[kt][c16] = f;
    }

  for (int mt = 0; mt < 4; ++mt) {
    // Q A-fragment for this row tile (vector loads)
    int r = mt * 16 + nl;
    v16h aq = (r < NTOK) ? ld16(qkv + rowbase + (size_t)r * 768 + colq + kb)
                         : zero_v16h();
    // S = q k^T (4 tiles across keys)
    v8f s[4];
#pragma unroll
    for (int nt = 0; nt < 4; ++nt) {
      v8f z = {};
      s[nt] = wmma_f32(aq, bk[nt], z);
    }
    // scale + rel-pos bias + mask + softmax (rows live in 16-lane halves)
    const int rr0 = mt * 16 + halfsel * 8;
#pragma unroll
    for (int j = 0; j < 8; ++j) {
      int rq = rr0 + j;
      float vals[4];
      float rmax = -1e30f;
#pragma unroll
      for (int nt = 0; nt < 4; ++nt) {
        int n = nt * 16 + nl;
        float val = -1e30f;
        if (rq < NTOK && n < NTOK) {
          int qi = rq / WS7, qj = rq % WS7, ki = n / WS7, kj = n % WS7;
          int idx = (qi - ki + 6) * 13 + (qj - kj + 6);
          val = s[nt][j] * ATT_SCALE + rpb[idx * NHEAD + head];
        }
        vals[nt] = val;
        rmax = fmaxf(rmax, val);
      }
#pragma unroll
      for (int msk = 8; msk; msk >>= 1) rmax = fmaxf(rmax, __shfl_xor(rmax, msk, 32));
      float rs = 0.f;
#pragma unroll
      for (int nt = 0; nt < 4; ++nt) { vals[nt] = __expf(vals[nt] - rmax); rs += vals[nt]; }
#pragma unroll
      for (int msk = 8; msk; msk >>= 1) rs += __shfl_xor(rs, msk, 32);
      float inv = 1.f / rs;
#pragma unroll
      for (int nt = 0; nt < 4; ++nt)
        pl[head][(halfsel * 8 + j) * 72 + nt * 16 + nl] = (_Float16)(vals[nt] * inv);
    }
    // O_mt = P_mt (16x64) @ V (64x32), re-layout via LDS slice
    v8f o0 = {}, o1 = {};
#pragma unroll
    for (int kt = 0; kt < 2; ++kt) {
      v16h ap = frag_a(&pl[head][kt * 32], 72, lane);
      o0 = wmma_f32(ap, bv[kt][0], o0);
      o1 = wmma_f32(ap, bv[kt][1], o1);
    }
#pragma unroll
    for (int j = 0; j < 8; ++j) {
      int rq = rr0 + j;
      if (rq < NTOK) {
        size_t orow = ((size_t)win * NTOK + rq) * CDIM + head * HDIM;
        obuf[orow + nl]      = (_Float16)o0[j];
        obuf[orow + 16 + nl] = (_Float16)o1[j];
      }
    }
  }
}

// ---------------- Kernel: mean pool over 196 tokens + classifier (256 -> 7) ----------------
__global__ __launch_bounds__(256) void k_pool_cls(const float* __restrict__ tok3,
                                                  const float* __restrict__ w,
                                                  const float* __restrict__ bias,
                                                  float* __restrict__ out) {
  __shared__ float pool[CDIM];
  const int b = blockIdx.x, c = threadIdx.x;
  const float* p = tok3 + (size_t)b * HW * CDIM + c;
  float s = 0.f;
  for (int t = 0; t < HW; ++t) s += p[(size_t)t * CDIM];
  pool[c] = s * (1.f / (float)HW);
  __syncthreads();
  if (c < 7) {
    float acc = bias[c];
    for (int k = 0; k < CDIM; ++k) acc += pool[k] * w[k * 7 + c];
    out[b * 7 + c] = acc;
  }
}

// ---------------- Launch ----------------
extern "C" void kernel_launch(void* const* d_in, const int* in_sizes, int n_in,
                              void* d_out, int out_size, void* d_ws, size_t ws_size,
                              hipStream_t stream) {
  const float* x      = (const float*)d_in[0];
  const float* conv_w = (const float*)d_in[1];
  const float* conv_b = (const float*)d_in[2];
  const float* ln1_g  = (const float*)d_in[3];
  const float* ln1_b  = (const float*)d_in[4];
  const float* qkv_w  = (const float*)d_in[5];
  const float* qkv_b  = (const float*)d_in[6];
  const float* rpb    = (const float*)d_in[7];
  const float* proj_w = (const float*)d_in[8];
  const float* proj_b = (const float*)d_in[9];
  const float* ln2_g  = (const float*)d_in[10];
  const float* ln2_b  = (const float*)d_in[11];
  const float* fc1_w  = (const float*)d_in[12];
  const float* fc1_b  = (const float*)d_in[13];
  const float* fc2_w  = (const float*)d_in[14];
  const float* fc2_b  = (const float*)d_in[15];
  const float* fco_w  = (const float*)d_in[16];
  const float* fco_b  = (const float*)d_in[17];

  char* ws = (char*)d_ws;
  constexpr size_t SZ_F32 = (size_t)MROWS * CDIM * 4;   // 51,380,224
  constexpr size_t SZ_F16 = (size_t)MROWS * CDIM * 2;   // 25,690,112
  constexpr size_t SZ_QKV = (size_t)MROWS * 768 * 2;    // 77,070,336
  constexpr size_t SZ_G16 = (size_t)MROWS * 1024 * 2;   // 102,760,448
  float*    tok   = (float*)(ws);
  float*    tok2  = (float*)(ws + SZ_F32);
  float*    tok3  = (float*)(ws + 2 * SZ_F32);
  _Float16* yw    = (_Float16*)(ws + 3 * SZ_F32);
  _Float16* qkv16 = (_Float16*)(ws + 3 * SZ_F32 + SZ_F16);
  _Float16* obuf  = (_Float16*)(ws + 3 * SZ_F32 + SZ_F16 + SZ_QKV);
  _Float16* h16   = (_Float16*)(ws + 3 * SZ_F32 + 2 * SZ_F16 + SZ_QKV);
  _Float16* g16   = (_Float16*)(ws + 3 * SZ_F32 + 3 * SZ_F16 + SZ_QKV);
  char*     wb    = ws + 3 * SZ_F32 + 3 * SZ_F16 + SZ_QKV + SZ_G16;
  _Float16* convW16 = (_Float16*)(wb);                      // 256 x 256  (N x K)
  _Float16* qkvW16  = (_Float16*)(wb + 2 * 65536);          // 768 x 256
  _Float16* projW16 = (_Float16*)(wb + 2 * 65536 + 2 * 196608);
  _Float16* fc1W16  = (_Float16*)(wb + 2 * 65536 + 2 * 196608 + 2 * 65536);
  _Float16* fc2W16  = (_Float16*)(wb + 2 * 65536 + 2 * 196608 + 2 * 65536 + 2 * 262144);

  const dim3 blk(256);
  // 0. weight prep: f16, N x K layouts
  k_cvt<<<256, blk, 0, stream>>>(conv_w, convW16, 65536);            // already (o,c)=NxK
  k_cvt_t<<<768, blk, 0, stream>>>(qkv_w, qkvW16, 256, 768);
  k_cvt_t<<<256, blk, 0, stream>>>(proj_w, projW16, 256, 256);
  k_cvt_t<<<1024, blk, 0, stream>>>(fc1_w, fc1W16, 256, 1024);
  k_cvt_t<<<1024, blk, 0, stream>>>(fc2_w, fc2W16, 1024, 256);
  // 1. conv 1x1 -> tok (fp32, residual source)
  k_conv<<<dim3(392, 4), blk, 0, stream>>>(x, convW16, conv_b, tok);
  // 2. LN1 + window partition -> yw (f16, windowed row order)
  k_ln<<<6272, blk, 0, stream>>>(tok, ln1_g, ln1_b, yw, 1);
  // 3. qkv GEMM -> qkv16 (f16, 768 cols)
  {
    EpiF16 e{qkv16, 768};
    k_gemm<EpiF16><<<dim3(392, 12), blk, 0, stream>>>(yw, qkvW16, qkv_b, 256, e);
  }
  // 4. windowed attention -> obuf (f16, windowed rows, head-major cols)
  k_attn<<<1024, blk, 0, stream>>>(qkv16, rpb, obuf);
  // 5. proj GEMM + window reverse + residual -> tok2 (fp32)
  {
    EpiProj e{tok, tok2};
    k_gemm<EpiProj><<<dim3(392, 4), blk, 0, stream>>>(obuf, projW16, proj_b, 256, e);
  }
  // 6. LN2 -> h16 (f16)
  k_ln<<<6272, blk, 0, stream>>>(tok2, ln2_g, ln2_b, h16, 0);
  // 7. fc1 GEMM + GELU -> g16 (f16, 1024 cols)
  {
    EpiGelu e{g16, 1024};
    k_gemm<EpiGelu><<<dim3(392, 16), blk, 0, stream>>>(h16, fc1W16, fc1_b, 256, e);
  }
  // 8. fc2 GEMM + residual -> tok3 (fp32)
  {
    EpiFc2 e{tok2, tok3};
    k_gemm<EpiFc2><<<dim3(392, 4), blk, 0, stream>>>(g16, fc2W16, fc2_b, 1024, e);
  }
  // 9. mean pool + classifier -> out (256 x 7 fp32)
  k_pool_cls<<<256, blk, 0, stream>>>(tok3, fco_w, fco_b, (float*)d_out);
}
